// AttnDecoderRNN_69269232550542
// MI455X (gfx1250) — compile-verified, matrix operand order checked by reading, not run on hardware
//
#include <hip/hip_runtime.h>
#include <hip/hip_bf16.h>
#include <math.h>

typedef __attribute__((ext_vector_type(16))) _Float16 v16h;
typedef __attribute__((ext_vector_type(8)))  float    v8f;

#define BB    128
#define SS    256
#define VV    50000
#define EE    300
#define HH    300
#define ENC2  600
#define KATT  600
#define KATTP 608     // K padded to 19*32
#define NATTP 320     // N padded to 20*16 (processed as 10 pairs of tiles)
#define KOUT  1200
#define KOUTP 1216    // padded to 38*32

union FragH {
    v16h h;
    unsigned int u[8];
};

// pack two f32 into one dword of two f16 (lowers to v_cvt_pk_f16_f32)
__device__ __forceinline__ unsigned int pkh2(float a, float b) {
    union { unsigned int u; _Float16 h[2]; } t;
    t.h[0] = (_Float16)a;
    t.h[1] = (_Float16)b;
    return t.u;
}

// branch-free tanh: 1 - 2/(e^{2x}+1); saturates to +-1 on overflow/underflow
__device__ __forceinline__ float tanh_fast(float x) {
    float e = __expf(2.0f * x);
    return 1.0f - 2.0f * __builtin_amdgcn_rcpf(e + 1.0f);
}
// branch-free sigmoid
__device__ __forceinline__ float sigmoid_fast(float x) {
    return __builtin_amdgcn_rcpf(1.0f + __expf(-x));
}

// ---------------------------------------------------------------------------
// 0a) pack Wc = attn_W[300:900, 0:300] -> zero-padded TRANSPOSED f16 WcT[n][k]
//     (320 x 608), plus zero-padded attn_v (320).  One-time, tiny.
// ---------------------------------------------------------------------------
__global__ void pack_attn_kernel(const float* __restrict__ attn_W,
                                 const float* __restrict__ attn_v,
                                 _Float16* __restrict__ WcT,
                                 float* __restrict__ vpad) {
    int idx = blockIdx.x * blockDim.x + threadIdx.x;
    if (idx < NATTP * KATTP) {
        int n = idx / KATTP;
        int k = idx - n * KATTP;
        float v = (n < HH && k < KATT) ? attn_W[(size_t)(HH + k) * HH + n] : 0.0f;
        WcT[idx] = (_Float16)v;
    }
    if (idx < NATTP) vpad[idx] = (idx < HH) ? attn_v[idx] : 0.0f;
}

// ---------------------------------------------------------------------------
// 0b) stage context -> zero-padded f16 rows (stride 608).  One block per row,
//     coalesced float2 reads, packed dword stores.
// ---------------------------------------------------------------------------
__global__ void pack_context_kernel(const float* __restrict__ context,
                                    _Float16* __restrict__ ctxh) {
    int r = blockIdx.x;                       // 0 .. B*S-1
    const float* src = context + (size_t)r * ENC2;
    unsigned int* dst = (unsigned int*)(ctxh + (size_t)r * KATTP);
    for (int q = threadIdx.x; q < KATTP / 2; q += blockDim.x) {
        int k = 2 * q;
        unsigned int v = 0u;
        if (k < KATT) {
            float2 f = *(const float2*)(src + k);
            v = pkh2(f.x, f.y);
        }
        dst[q] = v;
    }
}

// ---------------------------------------------------------------------------
// 1) embedding gather + Hpart[b][n] = attn_b[n] + hidden[b] . attn_W[0:300, n]
//    Hpart stored zero-padded with stride 320.
// ---------------------------------------------------------------------------
__global__ void prep_kernel(const int* __restrict__ inputs,
                            const float* __restrict__ emb_table,
                            const float* __restrict__ hidden,
                            const float* __restrict__ attn_W,
                            const float* __restrict__ attn_b,
                            float* __restrict__ embedded,
                            float* __restrict__ Hpart) {
    int b = blockIdx.x;
    __shared__ float hrow[HH];
    for (int k = threadIdx.x; k < HH; k += blockDim.x) hrow[k] = hidden[b * HH + k];
    __syncthreads();
    int tok = inputs[b];
    for (int e = threadIdx.x; e < EE; e += blockDim.x)
        embedded[b * EE + e] = emb_table[(size_t)tok * EE + e];
    for (int n = threadIdx.x; n < NATTP; n += blockDim.x) {
        float s = 0.0f;
        if (n < HH) {
            s = attn_b[n];
            for (int k = 0; k < HH; ++k) s += hrow[k] * attn_W[(size_t)k * HH + n];
        }
        Hpart[b * NATTP + n] = s;
    }
}

// ---------------------------------------------------------------------------
// 2) WMMA GEMM + fused epilogue:
//    scores[r] = sum_n tanh( ctx_row[r] . Wc[:,n] + Hpart[b][n] ) * v[n]
//    32768 rows -> 2048 M-tiles of 16, one wave per tile.  Two N-tiles per
//    pass share each A fragment (2 wmma per A load); fully branch-free.
// ---------------------------------------------------------------------------
__global__ void attn_scores_kernel(const _Float16* __restrict__ ctxh,
                                   const _Float16* __restrict__ WcT,
                                   const float* __restrict__ Hpart,
                                   const float* __restrict__ vpad,
                                   float* __restrict__ scores) {
    int lane = threadIdx.x & 31;
    int wave = threadIdx.x >> 5;
    int tile = blockIdx.x * 8 + wave;
    int row0 = tile * 16;
    const _Float16* ctxrow = ctxh + (size_t)(row0 + (lane & 15)) * KATTP;
    int ncol  = lane & 15;
    int khalf = (lane >> 4) << 3;   // +8 for upper half-wave
    float partial[8] = {0.f, 0.f, 0.f, 0.f, 0.f, 0.f, 0.f, 0.f};

    for (int np = 0; np < 10; ++np) {          // 10 pairs of N-tiles
        int n0 = np * 32 + ncol;
        int n1 = n0 + 16;
        const _Float16* bcol0 = WcT + (size_t)n0 * KATTP;
        const _Float16* bcol1 = WcT + (size_t)n1 * KATTP;
        v8f acc0 = {0.f, 0.f, 0.f, 0.f, 0.f, 0.f, 0.f, 0.f};
        v8f acc1 = {0.f, 0.f, 0.f, 0.f, 0.f, 0.f, 0.f, 0.f};
        for (int kt = 0; kt < 19; ++kt) {
            int k0 = kt * 32 + khalf;
            FragH a, b0, b1;
#pragma unroll
            for (int g = 0; g < 8; ++g) {
                int k   = k0 + ((g & 4) << 2) + ((g & 3) << 1);
                a.u[g]  = *(const unsigned int*)(ctxrow + k);
                b0.u[g] = *(const unsigned int*)(bcol0 + k);
                b1.u[g] = *(const unsigned int*)(bcol1 + k);
            }
            acc0 = __builtin_amdgcn_wmma_f32_16x16x32_f16(false, a.h, false, b0.h,
                                                          (short)0, acc0, false, false);
            acc1 = __builtin_amdgcn_wmma_f32_16x16x32_f16(false, a.h, false, b1.h,
                                                          (short)0, acc1, false, false);
        }
        float vn0 = vpad[n0];
        float vn1 = vpad[n1];
#pragma unroll
        for (int g = 0; g < 8; ++g) {
            int m  = g + ((lane >> 4) << 3);
            int bi = (row0 + m) >> 8;           // S == 256
            const float* hp = Hpart + bi * NATTP;
            partial[g] += tanh_fast(acc0[g] + hp[n0]) * vn0;
            partial[g] += tanh_fast(acc1[g] + hp[n1]) * vn1;
        }
    }
#pragma unroll
    for (int g = 0; g < 8; ++g) {
        float p = partial[g];
        p += __shfl_xor(p, 1, 32);
        p += __shfl_xor(p, 2, 32);
        p += __shfl_xor(p, 4, 32);
        p += __shfl_xor(p, 8, 32);
        partial[g] = p;
    }
    if ((lane & 15) == 0) {
#pragma unroll
        for (int g = 0; g < 8; ++g)
            scores[row0 + g + ((lane >> 4) << 3)] = partial[g];
    }
}

// ---------------------------------------------------------------------------
// 3) masked softmax over S (in place, one block per batch row)
// ---------------------------------------------------------------------------
__global__ void softmax_kernel(float* __restrict__ scores,
                               const unsigned char* __restrict__ mask) {
    int b = blockIdx.x;
    int s = threadIdx.x;            // blockDim.x == 256 == SS
    __shared__ float red[SS];
    float val = scores[b * SS + s];
    if (!mask[b * SS + s]) val = -1.0e9f;
    red[s] = val;
    __syncthreads();
    for (int off = 128; off > 0; off >>= 1) {
        if (s < off) red[s] = fmaxf(red[s], red[s + off]);
        __syncthreads();
    }
    float mx = red[0];
    __syncthreads();
    float e = __expf(val - mx);
    red[s] = e;
    __syncthreads();
    for (int off = 128; off > 0; off >>= 1) {
        if (s < off) red[s] += red[s + off];
        __syncthreads();
    }
    scores[b * SS + s] = e / red[0];
}

// ---------------------------------------------------------------------------
// 4) attn_applied[b][d] = sum_s w[b][s] * ctx[b][s][d]  (reads f16 copy)
// ---------------------------------------------------------------------------
__global__ void attn_apply_kernel(const float* __restrict__ weights,
                                  const _Float16* __restrict__ ctxh,
                                  float* __restrict__ attn_applied) {
    int b = blockIdx.x;
    __shared__ float w[SS];
    for (int s = threadIdx.x; s < SS; s += blockDim.x) w[s] = weights[b * SS + s];
    __syncthreads();
    const _Float16* ctx = ctxh + (size_t)b * SS * KATTP;
    for (int d = threadIdx.x; d < ENC2; d += blockDim.x) {
        float acc = 0.f;
        for (int s = 0; s < SS; ++s) acc += w[s] * (float)ctx[(size_t)s * KATTP + d];
        attn_applied[b * ENC2 + d] = acc;
    }
}

// ---------------------------------------------------------------------------
// 5) GRU cell + build f16 row Y[b] = [h_new | attn_applied | embedded | 0-pad]
// ---------------------------------------------------------------------------
__global__ void gru_kernel(const float* __restrict__ embedded,
                           const float* __restrict__ attn_applied,
                           const float* __restrict__ hidden,
                           const float* __restrict__ W_ih, const float* __restrict__ b_ih,
                           const float* __restrict__ W_hh, const float* __restrict__ b_hh,
                           float* __restrict__ h_new_out, _Float16* __restrict__ Yh) {
    int b = blockIdx.x;
    __shared__ float x[900];
    __shared__ float hr[HH];
    __shared__ float gi[900];
    __shared__ float gh[900];
    __shared__ float hn[HH];
    for (int k = threadIdx.x; k < 900; k += blockDim.x)
        x[k] = (k < EE) ? embedded[b * EE + k] : attn_applied[b * ENC2 + (k - EE)];
    for (int k = threadIdx.x; k < HH; k += blockDim.x) hr[k] = hidden[b * HH + k];
    __syncthreads();
    for (int j = threadIdx.x; j < 900; j += blockDim.x) {
        float s1 = b_ih[j];
        for (int k = 0; k < 900; ++k) s1 += x[k] * W_ih[(size_t)j * 900 + k];
        gi[j] = s1;
        float s2 = b_hh[j];
        for (int k = 0; k < HH; ++k) s2 += hr[k] * W_hh[(size_t)j * HH + k];
        gh[j] = s2;
    }
    __syncthreads();
    for (int j = threadIdx.x; j < HH; j += blockDim.x) {
        float r = sigmoid_fast(gi[j] + gh[j]);
        float z = sigmoid_fast(gi[HH + j] + gh[HH + j]);
        float n = tanh_fast(gi[2 * HH + j] + r * gh[2 * HH + j]);
        float h = (1.0f - z) * n + z * hr[j];
        hn[j] = h;
        h_new_out[b * HH + j] = h;
    }
    __syncthreads();
    for (int t = threadIdx.x; t < KOUTP; t += blockDim.x) {
        float v;
        if (t < HH)             v = hn[t];
        else if (t < HH + ENC2) v = attn_applied[b * ENC2 + (t - HH)];
        else if (t < KOUT)      v = embedded[b * EE + (t - HH - ENC2)];
        else                    v = 0.f;
        Yh[(size_t)b * KOUTP + t] = (_Float16)v;
    }
}

// ---------------------------------------------------------------------------
// 6) logits GEMM: (128 x 1216) x (1216 x 50000).  One block per 16-column
//    out_W tile staged through LDS as f16 (float4 reads, packed dword stores);
//    8 waves cover the 8 M-tiles; dword fragment loads; branch-free.
// ---------------------------------------------------------------------------
__global__ void logits_kernel(const _Float16* __restrict__ Yh,
                              const float* __restrict__ out_W,
                              const float* __restrict__ out_b,
                              float* __restrict__ logits) {
    __shared__ _Float16 Bs[16 * KOUTP];   // 38.9 KB
    int nt  = blockIdx.x;                 // 0..3124
    int tid = threadIdx.x;

    // stage 16 rows of out_W -> f16 LDS; k in [1200,1216) zero-padded
    for (int nl = 0; nl < 16; ++nl) {
        const float4* src = (const float4*)(out_W + (size_t)(nt * 16 + nl) * KOUT);
        unsigned int* dst = (unsigned int*)(Bs + nl * KOUTP);
        for (int q = tid; q < KOUT / 4; q += 256) {
            float4 f = src[q];
            dst[2 * q]     = pkh2(f.x, f.y);
            dst[2 * q + 1] = pkh2(f.z, f.w);
        }
        if (tid < (KOUTP - KOUT) / 2) dst[KOUT / 2 + tid] = 0u;
    }
    __syncthreads();

    int lane  = tid & 31;
    int wave  = tid >> 5;
    int m0    = wave * 16;
    int khalf = (lane >> 4) << 3;
    const _Float16* yrow = Yh + (size_t)(m0 + (lane & 15)) * KOUTP;
    const _Float16* brow = Bs + (size_t)(lane & 15) * KOUTP;
    v8f acc = {0.f, 0.f, 0.f, 0.f, 0.f, 0.f, 0.f, 0.f};
    for (int kt = 0; kt < 38; ++kt) {
        int k0 = kt * 32 + khalf;
        FragH a, bm;
#pragma unroll
        for (int g = 0; g < 8; ++g) {
            int k   = k0 + ((g & 4) << 2) + ((g & 3) << 1);
            a.u[g]  = *(const unsigned int*)(yrow + k);
            bm.u[g] = *(const unsigned int*)(brow + k);
        }
        acc = __builtin_amdgcn_wmma_f32_16x16x32_f16(false, a.h, false, bm.h,
                                                     (short)0, acc, false, false);
    }
    int n = nt * 16 + (lane & 15);
    float bias = out_b[n];
#pragma unroll
    for (int g = 0; g < 8; ++g) {
        int m = m0 + g + ((lane >> 4) << 3);
        logits[(size_t)m * VV + n] = acc[g] + bias;
    }
}

// ---------------------------------------------------------------------------
extern "C" void kernel_launch(void* const* d_in, const int* in_sizes, int n_in,
                              void* d_out, int out_size, void* d_ws, size_t ws_size,
                              hipStream_t stream) {
    (void)in_sizes; (void)n_in; (void)out_size; (void)ws_size;
    const int*           inputs    = (const int*)d_in[0];
    const float*         hidden    = (const float*)d_in[1];
    const float*         context   = (const float*)d_in[2];
    const unsigned char* mask      = (const unsigned char*)d_in[3];
    const float*         emb_table = (const float*)d_in[4];
    const float*         attn_W    = (const float*)d_in[5];
    const float*         attn_b    = (const float*)d_in[6];
    const float*         attn_v    = (const float*)d_in[7];
    const float*         W_ih      = (const float*)d_in[8];
    const float*         b_ih      = (const float*)d_in[9];
    const float*         W_hh      = (const float*)d_in[10];
    const float*         b_hh      = (const float*)d_in[11];
    const float*         out_W     = (const float*)d_in[12];
    const float*         out_b     = (const float*)d_in[13];

    // outputs: logits (B*V) | h_new (B*H) | attn_weights (B*S)
    float* logits = (float*)d_out;
    float* h_new  = logits + (size_t)BB * VV;
    float* attn_w = h_new + (size_t)BB * HH;

    char* ws = (char*)d_ws;
    float*    embedded     = (float*)(ws);                  // 153600 B
    float*    Hpart        = (float*)(ws + 153600);         // 128*320*4 = 163840 B
    float*    attn_applied = (float*)(ws + 317440);         // 307200 B
    _Float16* Yh           = (_Float16*)(ws + 624640);      // 128*1216*2 = 311296 B
    _Float16* WcT          = (_Float16*)(ws + 935936);      // 320*608*2 = 389120 B
    float*    vpad         = (float*)(ws + 1325056);        // 320*4 B
    _Float16* ctxh         = (_Float16*)(ws + 1326336);     // 32768*608*2 = 39.8 MB

    pack_attn_kernel<<<(NATTP * KATTP + 255) / 256, 256, 0, stream>>>(attn_W, attn_v,
                                                                      WcT, vpad);
    pack_context_kernel<<<BB * SS, 256, 0, stream>>>(context, ctxh);
    prep_kernel<<<BB, 256, 0, stream>>>(inputs, emb_table, hidden, attn_W, attn_b,
                                        embedded, Hpart);
    attn_scores_kernel<<<(BB * SS / 16) / 8, 256, 0, stream>>>(ctxh, WcT, Hpart,
                                                               vpad, attn_w);
    softmax_kernel<<<BB, SS, 0, stream>>>(attn_w, mask);
    attn_apply_kernel<<<BB, 256, 0, stream>>>(attn_w, ctxh, attn_applied);
    gru_kernel<<<BB, 256, 0, stream>>>(embedded, attn_applied, hidden,
                                       W_ih, b_ih, W_hh, b_hh, h_new, Yh);
    logits_kernel<<<VV / 16, 256, 0, stream>>>(Yh, out_W, out_b, logits);
}
